// NGEModel_45827301048521
// MI455X (gfx1250) — compile-verified, matrix operand order checked by reading, not run on hardware
//
#include <hip/hip_runtime.h>

// ---------------------------------------------------------------------------
// NGE model forward for MI455X (gfx1250, wave32).
// - GEMMs through v_wmma_f32_16x16x32_bf16 (f32 accumulate).
// - LDS staging via GLOBAL_LOAD_ASYNC_TO_LDS_B128 (ASYNCcnt-tracked DMA),
//   double-buffered so tile k+1 streams while tile k is in the matrix pipes.
// Dominant cost: logits GEMM [4096x50000, K=256] -> 819MB output write
// (~35us at 23.3 TB/s HBM); node_emb (51MB) stays L2-resident (192MB L2).
// ---------------------------------------------------------------------------

#define VOCAB 50000
#define D 256
#define E 4096
#define S 9
#define A 4
#define NUM_ORDERS 3

typedef __attribute__((ext_vector_type(16))) __bf16 v16bf;
typedef __attribute__((ext_vector_type(8)))  __bf16 v8bf;
typedef __attribute__((ext_vector_type(8)))  float  v8f;

union BF16x16 { v16bf v; v8bf h[2]; };

#if __has_builtin(__builtin_amdgcn_s_wait_asynccnt)
#define WAIT_ASYNC0() __builtin_amdgcn_s_wait_asynccnt(0)
#else
#define WAIT_ASYNC0() asm volatile("s_wait_asynccnt 0" ::: "memory")
#endif

// Async DMA: one b128 from global (64-bit vaddr) into LDS (32-bit wave-relative
// address in the VDST vgpr). Tracked by ASYNCcnt, no data VGPRs consumed.
__device__ __forceinline__ void async_b128_to_lds(const void* gptr, void* lds_generic) {
  unsigned lds = (unsigned)(uintptr_t)lds_generic;       // low 32 bits = LDS offset
  unsigned long long ga = (unsigned long long)(uintptr_t)gptr;
  asm volatile("global_load_async_to_lds_b128 %0, %1, off"
               :: "v"(lds), "v"(ga) : "memory");
}

// ---------------- utility kernels ----------------

// packed f32 -> bf16: 8 elements per thread (2x b128 load, 1x b128 store)
__global__ void k_f2bf(const float* __restrict__ in, __bf16* __restrict__ out, int n8) {
  int i = blockIdx.x * blockDim.x + threadIdx.x;
  if (i < n8) {
    const float4* p = reinterpret_cast<const float4*>(in + (size_t)i * 8);
    float4 a = p[0], b = p[1];
    v8bf r;
    r[0] = (__bf16)a.x; r[1] = (__bf16)a.y; r[2] = (__bf16)a.z; r[3] = (__bf16)a.w;
    r[4] = (__bf16)b.x; r[5] = (__bf16)b.y; r[6] = (__bf16)b.z; r[7] = (__bf16)b.w;
    *reinterpret_cast<v8bf*>(out + (size_t)i * 8) = r;
  }
}

// Wt[n,k] = W[k,n] as bf16 (so every GEMM operand is [rows, K] with K contiguous)
__global__ void k_wt(const float* __restrict__ W, __bf16* __restrict__ Wt) {
  int n = blockIdx.x;          // D blocks
  int k = threadIdx.x;         // D threads
  Wt[(size_t)n * D + k] = (__bf16)W[(size_t)k * D + n];
}

__global__ void k_gather(const float* __restrict__ emb, const int* __restrict__ ids,
                         float* __restrict__ seq) {
  int row = blockIdx.x;        // s*E + e, total S*E rows
  int c = threadIdx.x;         // 64 float4 chunks per row
  int id = ids[row];
  const float4* src = reinterpret_cast<const float4*>(emb + (size_t)id * D);
  float4* dst = reinterpret_cast<float4*>(seq + (size_t)row * D);
  dst[c] = src[c];
}

__global__ void k_init_var(float* __restrict__ var0, float* __restrict__ var1) {
  int i = blockIdx.x * blockDim.x + threadIdx.x;
  var0[i] = 1.0f;
  var1[i] = 1.0f;
}

// ---------------- encoder: substitution + masked pool + out-token + LN ----------------

__global__ __launch_bounds__(256)
void k_encoder(float* __restrict__ seq, const float* __restrict__ var1,
               const float* __restrict__ input_mask, const int* __restrict__ var_in_mask,
               const int* __restrict__ vib, const int* __restrict__ out_pos,
               const float* __restrict__ ln_g, const float* __restrict__ ln_b,
               __bf16* __restrict__ h_bf) {
  int e = blockIdx.x;          // edge
  int d = threadIdx.x;         // channel
  float pooled = 0.0f, denom = 0.0f, out_tok = 0.0f;
  int op = out_pos[e];
  #pragma unroll
  for (int s = 0; s < S; ++s) {
    size_t idx = (size_t)s * E + e;
    float v = seq[idx * D + d];
    if (var_in_mask[idx] >= 0) {
      int src = vib[idx];
      src = src < 0 ? 0 : (src > E - 1 ? E - 1 : src);
      v = var1[(size_t)src * D + d];
      seq[idx * D + d] = v;      // persist substitution across orders
    }
    float m = input_mask[idx];
    pooled += v * m;
    denom += m;
    if (s == op) out_tok = v;
  }
  float x = pooled / fmaxf(denom, 1.0f) + out_tok;
  __shared__ float s1[256], s2[256];
  s1[d] = x; s2[d] = x * x;
  __syncthreads();
  for (int off = 128; off > 0; off >>= 1) {
    if (d < off) { s1[d] += s1[d + off]; s2[d] += s2[d + off]; }
    __syncthreads();
  }
  float mu = s1[0] * (1.0f / D);
  float var = s2[0] * (1.0f / D) - mu * mu;
  float h = (x - mu) * rsqrtf(var + 1e-5f) * ln_g[d] + ln_b[d];
  h_bf[(size_t)e * D + d] = (__bf16)h;
}

// ---------------- WMMA bf16 GEMM: C[M,N] = act(A[M,K=256] @ Bt[N,K=256]^T + bias) ---------
// Block: 256 threads = 8 waves in a 4(M) x 2(N) arrangement.
// Block tile 64(M) x 128(N); each wave owns 16x64 via 4 accumulators.
// Double-buffered LDS fed by async-to-LDS DMA: each thread issues exactly
// 3 async b128 per K-tile (uniform), so s_wait_asynccnt 0 at loop top waits
// for the current tile only; tile k+1 streams during the WMMAs of tile k.
// MODE: 0 = none (f32 out), 1 = relu (bf16 out), 2 = sigmoid (f32 out)

template <int MODE>
__global__ __launch_bounds__(256)
void k_gemm(const __bf16* __restrict__ Am, const __bf16* __restrict__ Bt,
            const float* __restrict__ bias, float* __restrict__ Cf,
            __bf16* __restrict__ Cbf, int N, int ldc) {
  __shared__ __align__(16) __bf16 As[2][64 * 32];
  __shared__ __align__(16) __bf16 Bs[2][128 * 32];

  const int tid  = threadIdx.x;
  const int lane = tid & 31;
  const int wave = tid >> 5;
  const int wm = wave & 3;           // 0..3 -> M sub-block
  const int wn = wave >> 2;          // 0..1 -> N sub-block
  const int m_blk = blockIdx.y * 64;
  const int n_blk = blockIdx.x * 128;

  v8f acc[4] = {};
  const int half = lane >> 4;        // wave32: two 16-lane halves
  const int r    = lane & 15;

  // per-thread staging coordinates (3 b128 chunks per K-tile)
  const int a_row = tid >> 2, a_kc = tid & 3;
  int b_gn0 = n_blk + (tid >> 2);        if (b_gn0 >= N) b_gn0 = N - 1;
  int b_gn1 = n_blk + ((tid + 256) >> 2); if (b_gn1 >= N) b_gn1 = N - 1;
  const int b_kc = tid & 3;

  auto issue_tile = [&](int buf, int k0) {
    async_b128_to_lds(Am + (size_t)(m_blk + a_row) * 256 + k0 + a_kc * 8,
                      &As[buf][a_row * 32 + a_kc * 8]);
    async_b128_to_lds(Bt + (size_t)b_gn0 * 256 + k0 + b_kc * 8,
                      &Bs[buf][(tid >> 2) * 32 + b_kc * 8]);
    async_b128_to_lds(Bt + (size_t)b_gn1 * 256 + k0 + b_kc * 8,
                      &Bs[buf][((tid + 256) >> 2) * 32 + b_kc * 8]);
  };

  issue_tile(0, 0);

  for (int kt = 0; kt < 8; ++kt) {
    const int cur = kt & 1;
    WAIT_ASYNC0();                 // this wave's DMA for tile kt has landed
    __syncthreads();               // ... and everyone else's
    if (kt + 1 < 8) issue_tile(cur ^ 1, (kt + 1) * 32);

    // A fragment per ISA layout: lanes 0-15 hold K[0..8)+K[16..24),
    // lanes 16-31 hold K[8..16)+K[24..32)
    BF16x16 a;
    a.h[0] = *reinterpret_cast<const v8bf*>(&As[cur][(wm * 16 + r) * 32 + half * 8]);
    a.h[1] = *reinterpret_cast<const v8bf*>(&As[cur][(wm * 16 + r) * 32 + 16 + half * 8]);
    #pragma unroll
    for (int t = 0; t < 4; ++t) {
      BF16x16 b;
      int brow = wn * 64 + t * 16 + r;
      b.h[0] = *reinterpret_cast<const v8bf*>(&Bs[cur][brow * 32 + half * 8]);
      b.h[1] = *reinterpret_cast<const v8bf*>(&Bs[cur][brow * 32 + 16 + half * 8]);
      acc[t] = __builtin_amdgcn_wmma_f32_16x16x32_bf16(
          /*neg_a=*/false, a.v, /*neg_b=*/false, b.v,
          /*c_mod=*/(short)0, acc[t], /*reuse_a=*/false, /*reuse_b=*/false);
    }
    // no trailing barrier needed: each wave's LDS reads complete before its
    // WMMA issue (register deps force dscnt drain), which precedes its next
    // barrier arrival; the buffer is only overwritten two iterations later.
  }

  // epilogue: C/D layout -> VGPR i holds M = i (lanes 0-15) / M = 8+i (lanes 16-31)
  const int col = lane & 15;
  const int mrow0 = m_blk + wm * 16 + half * 8;
  #pragma unroll
  for (int t = 0; t < 4; ++t) {
    int gn = n_blk + wn * 64 + t * 16 + col;
    if (gn >= N) continue;
    float bv = (MODE != 0) ? bias[gn] : 0.0f;
    #pragma unroll
    for (int i = 0; i < 8; ++i) {
      float v = acc[t][i] + bv;
      size_t off = (size_t)(mrow0 + i) * ldc + gn;
      if (MODE == 1) {
        v = v > 0.0f ? v : 0.0f;
        Cbf[off] = (__bf16)v;
      } else {
        if (MODE == 2) v = 1.0f / (1.0f + __expf(-v));
        Cf[off] = v;
      }
    }
  }
}

// ---------------- product-logic kernels (JAX read-then-write semantics) ----------------

__global__ void k_scatter_enc(const float* __restrict__ enc, const int* __restrict__ orders,
                              const int* __restrict__ ltypes, float* __restrict__ var0,
                              float* __restrict__ var1, int order) {
  int i = blockIdx.x * blockDim.x + threadIdx.x;
  int e = i >> 8;
  if (orders[e] == order) {
    float v = enc[i];
    var0[i] = v;
    var1[i] = (ltypes[e] == 1) ? (1.0f - v) : v;   // fused negation
  }
}

// conjunction: reads var1, writes varT (copy-through elsewhere)
__global__ void k_conj(const float* __restrict__ var0, const float* __restrict__ var1,
                       float* __restrict__ varT, const int* __restrict__ orders,
                       const int* __restrict__ ltypes, const int* __restrict__ lin,
                       const int* __restrict__ lib, int order) {
  int i = blockIdx.x * blockDim.x + threadIdx.x;
  int e = i >> 8, d = i & 255;
  float v = var1[i];
  if (orders[e] == order && ltypes[e] == 2) {
    float p = var0[i];
    #pragma unroll
    for (int a = 1; a < A; ++a) {
      if (lin[a * E + e] >= 0) {
        int s = lib[a * E + e];
        s = s < 0 ? 0 : (s > E - 1 ? E - 1 : s);
        p *= var1[(size_t)s * D + d];
      }
    }
    v = p;
  }
  varT[i] = v;
}

// disjunction: reads varT (post-conj state), writes var1
__global__ void k_disj(const float* __restrict__ var0, const float* __restrict__ varT,
                       float* __restrict__ var1, const int* __restrict__ orders,
                       const int* __restrict__ ltypes, const int* __restrict__ lin,
                       const int* __restrict__ lib, int order) {
  int i = blockIdx.x * blockDim.x + threadIdx.x;
  int e = i >> 8, d = i & 255;
  float v = varT[i];
  if (orders[e] == order && ltypes[e] == 3) {
    float p = 1.0f - var0[i];
    #pragma unroll
    for (int a = 1; a < A; ++a) {
      if (lin[a * E + e] >= 0) {
        int s = lib[a * E + e];
        s = s < 0 ? 0 : (s > E - 1 ? E - 1 : s);
        p *= (1.0f - varT[(size_t)s * D + d]);
      }
    }
    v = 1.0f - p;
  }
  var1[i] = v;
}

// ---------------- soft-label cross entropy ----------------

__global__ __launch_bounds__(256)
void k_ce(const float* __restrict__ logits, const int* __restrict__ answers,
          const int* __restrict__ mask_type, float* __restrict__ rowloss) {
  int e = blockIdx.x, t = threadIdx.x;
  const float* row = logits + (size_t)e * VOCAB;
  __shared__ float sm[256], sm2[256];

  float lmax = -1e30f;
  for (int j = t; j < VOCAB; j += 256) lmax = fmaxf(lmax, row[j]);
  sm[t] = lmax; __syncthreads();
  for (int off = 128; off > 0; off >>= 1) {
    if (t < off) sm[t] = fmaxf(sm[t], sm[t + off]);
    __syncthreads();
  }
  float mx = sm[0];
  __syncthreads();

  float se = 0.0f, sx = 0.0f;
  for (int j = t; j < VOCAB; j += 256) {
    float v = row[j];
    se += __expf(v - mx);
    sx += v;
  }
  sm[t] = se; sm2[t] = sx; __syncthreads();
  for (int off = 128; off > 0; off >>= 1) {
    if (t < off) { sm[t] += sm[t + off]; sm2[t] += sm2[t + off]; }
    __syncthreads();
  }
  if (t == 0) {
    float logZ = mx + logf(sm[0]);
    float sum_lp = sm2[0] - (float)VOCAB * logZ;
    float la = row[answers[e]] - logZ;
    float soft = (mask_type[e] > 0) ? 0.9f : 1.0f;
    rowloss[e] = -(soft * la + (1.0f - soft) / (float)(VOCAB - 1) * (sum_lp - la));
  }
}

__global__ __launch_bounds__(256)
void k_loss_final(const float* __restrict__ rowloss, float* __restrict__ out) {
  __shared__ float sm[256];
  float s = 0.0f;
  for (int i = threadIdx.x; i < E; i += 256) s += rowloss[i];
  sm[threadIdx.x] = s; __syncthreads();
  for (int off = 128; off > 0; off >>= 1) {
    if (threadIdx.x < off) sm[threadIdx.x] += sm[threadIdx.x + off];
    __syncthreads();
  }
  if (threadIdx.x == 0) out[0] = sm[0] * (1.0f / E);
}

// ---------------- host orchestration ----------------

extern "C" void kernel_launch(void* const* d_in, const int* in_sizes, int n_in,
                              void* d_out, int out_size, void* d_ws, size_t ws_size,
                              hipStream_t stream) {
  (void)in_sizes; (void)n_in; (void)out_size; (void)ws_size;
  const float* node_emb   = (const float*)d_in[0];
  const int*   input_ids  = (const int*)d_in[1];
  const float* input_mask = (const float*)d_in[2];
  const int*   edge_orders= (const int*)d_in[3];
  const int*   var_in_mask= (const int*)d_in[4];
  const int*   vib        = (const int*)d_in[5];
  const int*   out_pos    = (const int*)d_in[6];
  const int*   lin        = (const int*)d_in[7];
  const int*   lib        = (const int*)d_in[8];
  const int*   ltypes     = (const int*)d_in[9];
  const int*   answers    = (const int*)d_in[10];
  const int*   mask_type  = (const int*)d_in[11];
  const float* W1 = (const float*)d_in[12];
  const float* b1 = (const float*)d_in[13];
  const float* W2 = (const float*)d_in[14];
  const float* b2 = (const float*)d_in[15];
  const float* ln_g = (const float*)d_in[16];
  const float* ln_b = (const float*)d_in[17];

  float* out = (float*)d_out;
  float* logits = out + 1;                    // d_out = [loss, logits...]

  // workspace carving (~87 MB total)
  char* p = (char*)d_ws;
  auto carve = [&](size_t bytes) { char* r = p; p += (bytes + 255) & ~(size_t)255; return r; };
  float*  seq     = (float*) carve(sizeof(float) * S * E * D);
  __bf16* emb_bf  = (__bf16*)carve(sizeof(__bf16) * (size_t)VOCAB * D);
  __bf16* w1t     = (__bf16*)carve(sizeof(__bf16) * D * D);
  __bf16* w2t     = (__bf16*)carve(sizeof(__bf16) * D * D);
  __bf16* h_bf    = (__bf16*)carve(sizeof(__bf16) * E * D);
  __bf16* mid_bf  = (__bf16*)carve(sizeof(__bf16) * E * D);
  float*  enc     = (float*) carve(sizeof(float) * E * D);
  float*  var0    = (float*) carve(sizeof(float) * E * D);
  float*  var1    = (float*) carve(sizeof(float) * E * D);
  float*  varT    = (float*) carve(sizeof(float) * E * D);
  __bf16* var1_bf = (__bf16*)carve(sizeof(__bf16) * E * D);
  float*  rowloss = (float*) carve(sizeof(float) * E);

  // prep
  k_f2bf<<<(VOCAB * D / 8 + 255) / 256, 256, 0, stream>>>(node_emb, emb_bf, VOCAB * D / 8);
  k_wt  <<<D, D, 0, stream>>>(W1, w1t);
  k_wt  <<<D, D, 0, stream>>>(W2, w2t);
  k_gather<<<S * E, 64, 0, stream>>>(node_emb, input_ids, seq);
  k_init_var<<<E * D / 256, 256, 0, stream>>>(var0, var1);

  for (int order = 0; order < NUM_ORDERS; ++order) {
    k_encoder<<<E, D, 0, stream>>>(seq, var1, input_mask, var_in_mask, vib,
                                   out_pos, ln_g, ln_b, h_bf);
    // mid = relu(h @ W1 + b1) ; enc = sigmoid(mid @ W2 + b2)
    k_gemm<1><<<dim3(2, 64), 256, 0, stream>>>(h_bf, w1t, b1, nullptr, mid_bf, D, D);
    k_gemm<2><<<dim3(2, 64), 256, 0, stream>>>(mid_bf, w2t, b2, enc, nullptr, D, D);
    k_scatter_enc<<<E * D / 256, 256, 0, stream>>>(enc, edge_orders, ltypes,
                                                   var0, var1, order);
    k_conj<<<E * D / 256, 256, 0, stream>>>(var0, var1, varT, edge_orders, ltypes,
                                            lin, lib, order);
    k_disj<<<E * D / 256, 256, 0, stream>>>(var0, varT, var1, edge_orders, ltypes,
                                            lin, lib, order);
  }

  // logits = var1 @ node_emb.T  (the 105 GFLOP / 819 MB-write workhorse)
  k_f2bf<<<(E * D / 8 + 255) / 256, 256, 0, stream>>>(var1, var1_bf, E * D / 8);
  k_gemm<0><<<dim3((VOCAB + 127) / 128, 64), 256, 0, stream>>>(
      var1_bf, emb_bf, nullptr, logits, nullptr, VOCAB, VOCAB);

  // loss
  k_ce<<<E, 256, 0, stream>>>(logits, answers, mask_type, rowloss);
  k_loss_final<<<1, 256, 0, stream>>>(rowloss, out);
}
